// TensorRT_kernel_76536317215264
// MI455X (gfx1250) — matrix operand order repaired and checked by reading, not compiled, not run on hardware
//
#include <hip/hip_runtime.h>

// ---------------------------------------------------------------------------
// Shapes (fixed by the reference)
// ---------------------------------------------------------------------------
#define SEQ    1024
#define HDIM   4096
#define NH     32
#define DH     128
#define CTX    4096
#define KVLEN  (CTX + SEQ)          // 5120
#define NCHUNK (KVLEN / 32)         // 160 kv chunks of 32

typedef __attribute__((ext_vector_type(16))) __bf16 v16bf;
typedef __attribute__((ext_vector_type(8)))  float  v8f;

union Frag16 { v16bf v; uint4 u[2]; };

// Build a 16-element bf16 fragment from two 16-byte segments (A-frag pattern).
__device__ inline v16bf load_frag2(const __bf16* p0, const __bf16* p1) {
  Frag16 f;
  f.u[0] = *(const uint4*)p0;
  f.u[1] = *(const uint4*)p1;
  return f.v;
}

__device__ inline unsigned short bf16_bits(float x) {
  union { __bf16 b; unsigned short s; } u;
  u.b = (__bf16)x;
  return u.s;
}

// ---------------------------------------------------------------------------
// Kernel 1: fp32 -> bf16 convert (grid-stride)
// ---------------------------------------------------------------------------
__global__ void cvt_f32_bf16(const float* __restrict__ in, __bf16* __restrict__ out,
                             size_t n) {
  size_t i = (size_t)blockIdx.x * blockDim.x + threadIdx.x;
  size_t stride = (size_t)gridDim.x * blockDim.x;
  for (; i < n; i += stride) out[i] = (__bf16)in[i];
}

// ---------------------------------------------------------------------------
// Kernel 2: K cache [pos][head][d] fp32 -> head-major bf16 Kf[h][pos][d]
// (positions 0..4095; GEMM fills positions 4096..5119)
// ---------------------------------------------------------------------------
__global__ void build_kcache(const float* __restrict__ cK, __bf16* __restrict__ Kf) {
  size_t i = (size_t)blockIdx.x * blockDim.x + threadIdx.x;
  size_t stride = (size_t)gridDim.x * blockDim.x;
  const size_t n = (size_t)NH * CTX * DH;
  for (; i < n; i += stride) {
    int d   = (int)(i & (DH - 1));
    int pos = (int)((i >> 7) & (CTX - 1));
    int h   = (int)(i >> 19);
    Kf[((size_t)h * KVLEN + pos) * DH + d] = (__bf16)cK[((size_t)pos * NH + h) * DH + d];
  }
}

// ---------------------------------------------------------------------------
// Kernel 3: QKV GEMM  C[1024,12288] = Xb * Wb^T  via v_wmma_f32_16x16x32_bf16
// Workgroup = 256 threads (8 waves), tile 128x256, each wave 64x64 (16 C tiles).
// A/B fragments loaded straight from global in exact WMMA register layouts.
// Epilogue scatters columns into Q (pre-scaled by log2e/sqrt(d)), K-new, V-new.
// ---------------------------------------------------------------------------
__global__ __launch_bounds__(256) void qkv_gemm(
    const __bf16* __restrict__ Xb, const __bf16* __restrict__ Wb,
    __bf16* __restrict__ Qb, __bf16* __restrict__ Kf, __bf16* __restrict__ Vn) {
  const int lane = threadIdx.x & 31;
  const int wave = threadIdx.x >> 5;
  const int wr = wave >> 2, wc = wave & 3;
  const int m0 = blockIdx.x * 128 + wr * 64;
  const int n0 = blockIdx.y * 256 + wc * 64;
  const int half = lane >> 4;       // 0: lanes 0-15, 1: lanes 16-31
  const int q16  = lane & 15;

  v8f acc[4][4] = {};

  const int aoff = half * 8;        // A-frag: k+{0..7,16..23} vs k+{8..15,24..31}
  const int boff = half * 8;        // B-frag: k+{0..7,16..23} vs k+{8..15,24..31}

  for (int k = 0; k < HDIM; k += 32) {
    v16bf a[4], b[4];
#pragma unroll
    for (int i = 0; i < 4; ++i) {
      const __bf16* ap = Xb + (size_t)(m0 + 16 * i + q16) * HDIM + k + aoff;
      a[i] = load_frag2(ap, ap + 16);
    }
#pragma unroll
    for (int j = 0; j < 4; ++j) {
      const __bf16* bp = Wb + (size_t)(n0 + 16 * j + q16) * HDIM + k + boff;
      b[j] = load_frag2(bp, bp + 16);
    }
#pragma unroll
    for (int i = 0; i < 4; ++i)
#pragma unroll
      for (int j = 0; j < 4; ++j)
        acc[i][j] = __builtin_amdgcn_wmma_f32_16x16x32_bf16(
            false, a[i], false, b[j], (short)0, acc[i][j], false, false);
  }

  // softmax scale folded into Q, in log2 domain: (1/sqrt(128)) * log2(e)
  const float qscale = 0.088388347652310225f * 1.4426950408889634f;
#pragma unroll
  for (int j = 0; j < 4; ++j) {
    const int ncol = n0 + 16 * j;
    const int sec  = ncol >> 12;            // 0=Q, 1=K, 2=V (4096-wide sections)
    const int cis  = ncol & 4095;
    const int h    = cis >> 7;
    const int d    = (cis & 127) + q16;
#pragma unroll
    for (int i = 0; i < 4; ++i) {
#pragma unroll
      for (int r = 0; r < 8; ++r) {
        const int m = m0 + 16 * i + r + 8 * half;
        const float v = acc[i][j][r];
        if (sec == 0)
          Qb[((size_t)h * SEQ + m) * DH + d] = (__bf16)(v * qscale);
        else if (sec == 1)
          Kf[((size_t)h * KVLEN + CTX + m) * DH + d] = (__bf16)v;
        else
          Vn[((size_t)h * SEQ + m) * DH + d] = (__bf16)v;
      }
    }
  }
}

// ---------------------------------------------------------------------------
// Kernel 4: swizzle V (fp32 cache rows + bf16 new rows) into exact WMMA
// B-fragment order: Vsw[h][chunk][dtile][lane][j] = pack(V[kv0], V[kv0+1])
// with kv0 = 32*chunk + 2*j + 8*(j>=4) + 8*(lane>=16), d = 16*dtile + (lane&15).
// ---------------------------------------------------------------------------
__global__ void build_vsw(const float* __restrict__ cV, const __bf16* __restrict__ Vn,
                          unsigned int* __restrict__ Vsw) {
  size_t i = (size_t)blockIdx.x * blockDim.x + threadIdx.x;
  size_t stride = (size_t)gridDim.x * blockDim.x;
  const size_t n = (size_t)NH * NCHUNK * 8 * 32 * 8;
  for (; i < n; i += stride) {
    size_t t0 = i;
    const int j    = (int)(t0 & 7);  t0 >>= 3;
    const int lane = (int)(t0 & 31); t0 >>= 5;
    const int dt   = (int)(t0 & 7);  t0 >>= 3;
    const int c    = (int)(t0 % NCHUNK);
    const int h    = (int)(t0 / NCHUNK);
    const int d    = 16 * dt + (lane & 15);
    const int kv0  = 32 * c + 2 * j + ((j >> 2) << 3) + ((lane >> 4) << 3);
    float v0, v1;
    if (kv0 < CTX) {
      v0 = cV[((size_t)kv0 * NH + h) * DH + d];
      v1 = cV[((size_t)(kv0 + 1) * NH + h) * DH + d];
    } else {
      v0 = (float)Vn[((size_t)h * SEQ + (kv0 - CTX)) * DH + d];
      v1 = (float)Vn[((size_t)h * SEQ + (kv0 + 1 - CTX)) * DH + d];
    }
    Vsw[i] = (unsigned int)bf16_bits(v0) | ((unsigned int)bf16_bits(v1) << 16);
  }
}

// ---------------------------------------------------------------------------
// Kernel 5: flash attention. Grid (32 heads, 16), 128 threads = 4 waves,
// each wave owns one 16-query tile and streams 160 chunks of 32 keys.
// S' = K_chunk(32x128) . Q^T(128x16) puts one query per lane, and
// exp(S'-m) lands exactly in the A-operand layout for O += P.V.
// ---------------------------------------------------------------------------
__global__ __launch_bounds__(128) void flash_attn(
    const __bf16* __restrict__ Qb, const __bf16* __restrict__ Kf,
    const unsigned int* __restrict__ Vsw, float* __restrict__ out) {
  const int h    = blockIdx.x;
  const int lane = threadIdx.x & 31;
  const int wave = threadIdx.x >> 5;
  const int qbase = (blockIdx.y * 4 + wave) * 16;
  const int half = lane >> 4;
  const int q16  = lane & 15;

  // Q fragments (B operand of S'): lane = query, d {0..7,16..23} | {8..15,24..31} per lane half.
  const __bf16* qrow = Qb + ((size_t)h * SEQ + qbase + q16) * DH;
  v16bf Qf[4];
#pragma unroll
  for (int s = 0; s < 4; ++s)
    Qf[s] = load_frag2(qrow + 32 * s + 8 * half, qrow + 32 * s + 8 * half + 16);

  v8f O[8] = {};
  float mrun = -1e30f, lrun = 0.0f;

  const __bf16* Kh = Kf + (size_t)h * KVLEN * DH;
  const unsigned int* Vh = Vsw + (size_t)h * NCHUNK * 8 * 32 * 8;

  for (int c = 0; c < NCHUNK; ++c) {
    // ---- scores: S0 = rows kv 0..15, S1 = rows kv 16..31 (transposed) ----
    v8f S0 = {}, S1 = {};
    const __bf16* k0 = Kh + (size_t)(c * 32 + q16) * DH + 8 * half;
    const __bf16* k1 = k0 + 16 * DH;
#pragma unroll
    for (int s = 0; s < 4; ++s) {
      v16bf A0 = load_frag2(k0 + 32 * s, k0 + 32 * s + 16);
      v16bf A1 = load_frag2(k1 + 32 * s, k1 + 32 * s + 16);
      S0 = __builtin_amdgcn_wmma_f32_16x16x32_bf16(false, A0, false, Qf[s], (short)0, S0, false, false);
      S1 = __builtin_amdgcn_wmma_f32_16x16x32_bf16(false, A1, false, Qf[s], (short)0, S1, false, false);
    }

    // Warm L2/L0 for the next chunk's K rows while softmax VALU runs.
    if (c + 1 < NCHUNK)
      __builtin_prefetch(Kh + (size_t)((c + 1) * 32 + lane) * DH, 0, 3);

    // Hoist V-fragment loads so they are in flight during the softmax chain.
    Frag16 vf[8];
#pragma unroll
    for (int t = 0; t < 8; ++t) {
      const unsigned int* vp = Vh + (((size_t)c * 8 + t) * 32 + lane) * 8;
      vf[t].u[0] = *(const uint4*)(vp);
      vf[t].u[1] = *(const uint4*)(vp + 4);
    }

    // ---- online softmax (per lane = per query row half; merge halves) ----
    float cm = S0[0];
#pragma unroll
    for (int r = 1; r < 8; ++r) cm = fmaxf(cm, S0[r]);
#pragma unroll
    for (int r = 0; r < 8; ++r) cm = fmaxf(cm, S1[r]);
    cm = fmaxf(cm, __shfl_xor(cm, 16, 32));

    // Rescale only when the running max actually moves (rare after warmup):
    // saves 8 ds_bpermute + 64 v_mul per chunk in steady state.
    if (__any(cm > mrun)) {
      const float mnew  = fmaxf(mrun, cm);
      const float alpha = exp2f(mrun - mnew);
      lrun *= alpha;
      mrun  = mnew;
#pragma unroll
      for (int r = 0; r < 8; ++r) {
        const float ar = __shfl(alpha, r + 8 * half, 32);
#pragma unroll
        for (int t = 0; t < 8; ++t) O[t][r] = O[t][r] * ar;
      }
    }

    v16bf Pa;
    float ls = 0.0f;
#pragma unroll
    for (int r = 0; r < 8; ++r) {
      float p0 = exp2f(S0[r] - mrun);
      float p1 = exp2f(S1[r] - mrun);
      ls += p0 + p1;
      Pa[r]     = (__bf16)p0;   // A-frag K = {0..7 | 8..15} by lane half
      Pa[8 + r] = (__bf16)p1;   // A-frag K = {16..23 | 24..31} by lane half
    }
    lrun += ls;

    // ---- O += P . V : V fragments pre-swizzled into exact B layout ----
#pragma unroll
    for (int t = 0; t < 8; ++t)
      O[t] = __builtin_amdgcn_wmma_f32_16x16x32_bf16(false, Pa, false, vf[t].v, (short)0, O[t], false, false);
  }

  // ---- normalize and store fp32 output [seq][h*128+d] ----
  const float ltot = lrun + __shfl_xor(lrun, 16, 32);
  const float inv  = 1.0f / ltot;
#pragma unroll
  for (int r = 0; r < 8; ++r) {
    const float ir = __shfl(inv, r + 8 * half, 32);
    const int row  = qbase + r + 8 * half;
#pragma unroll
    for (int t = 0; t < 8; ++t)
      out[(size_t)row * HDIM + h * DH + 16 * t + q16] = O[t][r] * ir;
  }
}

// ---------------------------------------------------------------------------
// Host side: workspace layout (200 MiB total) and kernel sequencing.
// ---------------------------------------------------------------------------
extern "C" void kernel_launch(void* const* d_in, const int* in_sizes, int n_in,
                              void* d_out, int out_size, void* d_ws, size_t ws_size,
                              hipStream_t stream) {
  const float* X  = (const float*)d_in[0];
  const float* W  = (const float*)d_in[1];
  const float* cK = (const float*)d_in[2];
  const float* cV = (const float*)d_in[3];
  float* out = (float*)d_out;

  char* ws = (char*)d_ws;
  __bf16* Xb = (__bf16*)(ws);                           //   8 MiB: X bf16
  __bf16* Wb = (__bf16*)(ws + (size_t)8   * (1 << 20)); //  96 MiB: W bf16
  __bf16* Qb = (__bf16*)(ws + (size_t)104 * (1 << 20)); //   8 MiB: Q (scaled), head-major
  __bf16* Kf = (__bf16*)(ws + (size_t)112 * (1 << 20)); //  40 MiB: K full, head-major
  __bf16* Vn = (__bf16*)(ws + (size_t)152 * (1 << 20)); //   8 MiB: V new rows
  unsigned int* Vsw = (unsigned int*)(ws + (size_t)160 * (1 << 20)); // 40 MiB

  cvt_f32_bf16<<<2048, 256, 0, stream>>>(X, Xb, (size_t)SEQ * HDIM);
  cvt_f32_bf16<<<8192, 256, 0, stream>>>(W, Wb, (size_t)3 * HDIM * HDIM);
  build_kcache<<<8192, 256, 0, stream>>>(cK, Kf);

  qkv_gemm<<<dim3(SEQ / 128, (3 * HDIM) / 256), 256, 0, stream>>>(Xb, Wb, Qb, Kf, Vn);

  build_vsw<<<8192, 256, 0, stream>>>(cV, Vn, Vsw);

  flash_attn<<<dim3(NH, SEQ / 64), 128, 0, stream>>>(Qb, Kf, Vsw, out);
}
